// ScaledDotProductAttention_8504035246473
// MI455X (gfx1250) — matrix operand order repaired and checked by reading, not compiled, not run on hardware
//
#include <hip/hip_runtime.h>
#include <hip/hip_bf16.h>

// ---------------------------------------------------------------------------
// Scaled dot-product attention for MI455X (gfx1250, wave32, WMMA).
//   scores = (Q K^T) * 1/sqrt(64) + other ; masked -> -1e9 ; softmax ;
//   context = attn @ V.  Outputs: context (B*H*S*D f32) then attn
//   (B*H*S*S f32), concatenated in d_out.
//
// One workgroup = one (b,h) x 16-query-row strip.  Full 16x2048 f32 score
// strip lives in LDS (~128KB of 320KB WGP LDS); both GEMMs run on
// v_wmma_f32_16x16x32_f16 with f32 accumulation.  Read-once (other/mask)
// and write-once (attn/ctx) streams use non-temporal hints so K/V stay
// L2-resident across the 128 blocks that share each (b,h) slice.
// ---------------------------------------------------------------------------

typedef __attribute__((ext_vector_type(16))) _Float16 v16h;
typedef __attribute__((ext_vector_type(8)))  float    v8f;
typedef __attribute__((ext_vector_type(4)))  float    v4f;

#define BATCH 2
#define HEADS 16
#define SEQ   2048
#define HDIM  64
#define NEGV  (-1.0e9f)
#define SCALE 0.125f           // 1/sqrt(64)
#define SROW  2052             // padded LDS row stride (dwords); 2052/4 = 513 v4f
#define SROW4 (SROW / 4)

__global__ __launch_bounds__(128, 1)
void sdpa_wmma_kernel(const float* __restrict__ Q,
                      const float* __restrict__ K,
                      const float* __restrict__ V,
                      const unsigned char* __restrict__ mask,
                      const float* __restrict__ other,
                      float* __restrict__ ctx,
                      float* __restrict__ attn)
{
    extern __shared__ char smem[];
    float*    sc     = (float*)smem;                         // [16][SROW] score strip
    _Float16* qt     = (_Float16*)(sc + 16 * SROW);          // [16][64] Q tile (f16)
    float*    red    = (float*)(qt + 16 * HDIM);             // [16][8] reduction scratch
    float*    rowmax = red + 16 * 8;                         // [16]
    float*    rowinv = rowmax + 16;                          // [16]

    const int bh  = blockIdx.y;          // 0..31 (b*H + h)
    const int q0  = blockIdx.x * 16;     // first query row of this strip
    const int tid = threadIdx.x;
    const int lane = tid & 31;
    const int wid  = tid >> 5;           // 0..3 (4 wave32s)

    const size_t qkvBase = (size_t)bh * SEQ * HDIM;
    const float* Qb = Q + qkvBase + (size_t)q0 * HDIM;
    const float* Kb = K + qkvBase;
    const float* Vb = V + qkvBase;
    const size_t ssBase = (size_t)bh * SEQ * SEQ + (size_t)q0 * SEQ;
    const float* otherb = other + ssBase;
    const unsigned char* maskb = mask + ssBase;

    // ---- load Q tile, convert to f16 in LDS ------------------------------
    #pragma unroll
    for (int i = 0; i < 8; ++i) {
        const int idx = tid + i * 128;                       // 16*64 = 1024 elems
        qt[idx] = (_Float16)Qb[idx];
    }
    __syncthreads();

    // ---- per-wave A fragments of Q (16x32 f16, two K-chunks of 32) -------
    const int khalf = (lane >> 4) & 1;   // which K-half this lane holds
    const int arow  = lane & 15;         // A-matrix row for this lane
    const int rowH  = (lane >> 4) << 3;  // C/D row offset (+8 for upper half-wave)
    const int nIdx  = lane & 15;         // C/D column within tile

    v16h aq0, aq1;
    #pragma unroll
    for (int e = 0; e < 16; ++e) {
        const int k = khalf * 8 + e + ((e < 8) ? 0 : 8);     // ISA 16-bit A layout
        aq0[e] = qt[arow * HDIM + k];
        aq1[e] = qt[arow * HDIM + 32 + k];
    }

    // ======================================================================
    // Phase 1: scores = Q K^T * scale + other, masked; into LDS.
    // Wave handles key tiles kt = wid, wid+4, ... (32 tiles of 16 keys).
    // Two independent accumulators so the two WMMAs pipeline.
    // ======================================================================
    for (int kt = wid; kt < SEQ / 16; kt += 4) {
        const float* Krow = Kb + (size_t)kt * 16 * HDIM;
        __builtin_prefetch(Krow + 4 * 16 * HDIM + lane, 0, 1);

        // B fragments: B[k][n] = K[key n][dim k]; lane%16 = n (key), element = k (table order)
        v16h b0, b1;
        #pragma unroll
        for (int e = 0; e < 16; ++e) {
            const int k = khalf * 8 + e + ((e < 8) ? 0 : 8);
            b0[e] = (_Float16)Krow[nIdx * HDIM + k];         // dims 0..31
            b1[e] = (_Float16)Krow[nIdx * HDIM + 32 + k];    // dims 32..63
        }

        v8f c0 = {}, c1 = {};
        c0 = __builtin_amdgcn_wmma_f32_16x16x32_f16(false, aq0, false, b0,
                                                    (short)0, c0, false, false);
        c1 = __builtin_amdgcn_wmma_f32_16x16x32_f16(false, aq1, false, b1,
                                                    (short)0, c1, false, false);

        const int colb = kt * 16 + nIdx;
        #pragma unroll
        for (int r = 0; r < 8; ++r) {
            const int row = r + rowH;
            const size_t gi = (size_t)row * SEQ + colb;
            float v = (c0[r] + c1[r]) * SCALE
                    + __builtin_nontemporal_load(&otherb[gi]);
            if (__builtin_nontemporal_load(&maskb[gi])) v = NEGV;
            sc[row * SROW + colb] = v;
        }
    }
    __syncthreads();

    // ======================================================================
    // Phase 2: row softmax over the 16x2048 LDS strip (vectorized 16B).
    // 16 rows x 8 threads; each thread scans 64 v4f at stride 8.
    // ======================================================================
    const int srow = tid >> 3;           // 0..15
    const int ssub = tid & 7;            // 0..7
    v4f* sc4 = (v4f*)sc;                 // row stride SROW4 = 513 v4f

    // pass A: row max
    {
        float m = -3.0e38f;
        #pragma unroll 4
        for (int j = 0; j < SEQ / 32; ++j) {
            const v4f f = sc4[srow * SROW4 + ssub + j * 8];
            m = fmaxf(m, fmaxf(fmaxf(f.x, f.y), fmaxf(f.z, f.w)));
        }
        red[srow * 8 + ssub] = m;
    }
    __syncthreads();
    if (tid < 16) {
        float m = red[tid * 8];
        #pragma unroll
        for (int j = 1; j < 8; ++j) m = fmaxf(m, red[tid * 8 + j]);
        rowmax[tid] = m;
    }
    __syncthreads();

    // pass B: exp + row sum (exp values written back to LDS)
    {
        const float m = rowmax[srow];
        float s = 0.0f;
        #pragma unroll 4
        for (int j = 0; j < SEQ / 32; ++j) {
            const int a = srow * SROW4 + ssub + j * 8;
            v4f f = sc4[a];
            f.x = __expf(f.x - m);
            f.y = __expf(f.y - m);
            f.z = __expf(f.z - m);
            f.w = __expf(f.w - m);
            sc4[a] = f;
            s += f.x + f.y + f.z + f.w;
        }
        red[srow * 8 + ssub] = s;
    }
    __syncthreads();
    if (tid < 16) {
        float s = red[tid * 8];
        #pragma unroll
        for (int j = 1; j < 8; ++j) s += red[tid * 8 + j];
        rowinv[tid] = 1.0f / s;
    }
    __syncthreads();

    // pass C: normalize in LDS + coalesced non-temporal attn writeback
    {
        v4f* attn4 = (v4f*)(attn + ssBase);                  // 16B aligned
        #pragma unroll 4
        for (int i = 0; i < (16 * SEQ / 4) / 128; ++i) {     // 64 iters
            const int lin4 = tid + i * 128;
            const int row  = lin4 >> 9;                      // /512 v4f per row
            const int col4 = lin4 & 511;
            const float inv = rowinv[row];
            v4f p = sc4[row * SROW4 + col4];
            p.x *= inv; p.y *= inv; p.z *= inv; p.w *= inv;
            sc4[row * SROW4 + col4] = p;
            __builtin_nontemporal_store(p, &attn4[lin4]);
        }
    }
    __syncthreads();

    // ======================================================================
    // Phase 3: context = P(16x2048) @ V(2048x64).
    // Wave `wid` owns output columns [wid*16, wid*16+16); contracts over all
    // 2048 keys in chunks of 32.  Two independent accumulators (even/odd
    // chunks) so consecutive WMMAs pipeline.
    // ======================================================================
    v8f cacc0 = {}, cacc1 = {};
    for (int kt2 = 0; kt2 < SEQ / 32; kt2 += 2) {
        #pragma unroll
        for (int half = 0; half < 2; ++half) {
            const int kbase = (kt2 + half) * 32;

            // A fragment: two contiguous 8-float runs -> 4x ds_load_b128
            const v4f* scrow4 = (const v4f*)(sc + arow * SROW);
            const int  base4  = (kbase >> 2) + khalf * 2;
            const v4f a0 = scrow4[base4];
            const v4f a1 = scrow4[base4 + 1];
            const v4f a2 = scrow4[base4 + 4];
            const v4f a3 = scrow4[base4 + 5];
            v16h a;
            a[0]  = (_Float16)a0.x; a[1]  = (_Float16)a0.y;
            a[2]  = (_Float16)a0.z; a[3]  = (_Float16)a0.w;
            a[4]  = (_Float16)a1.x; a[5]  = (_Float16)a1.y;
            a[6]  = (_Float16)a1.z; a[7]  = (_Float16)a1.w;
            a[8]  = (_Float16)a2.x; a[9]  = (_Float16)a2.y;
            a[10] = (_Float16)a2.z; a[11] = (_Float16)a2.w;
            a[12] = (_Float16)a3.x; a[13] = (_Float16)a3.y;
            a[14] = (_Float16)a3.z; a[15] = (_Float16)a3.w;

            // B fragment: B[k][n] = V[kbase+k][wid*16+n]; lane%16 = n,
            // element = k (table order)
            v16h b;
            #pragma unroll
            for (int e = 0; e < 16; ++e) {
                const int k = khalf * 8 + e + ((e < 8) ? 0 : 8);
                b[e] = (_Float16)Vb[(size_t)(kbase + k) * HDIM + wid * 16 + nIdx];
            }

            if (half == 0)
                cacc0 = __builtin_amdgcn_wmma_f32_16x16x32_f16(
                            false, a, false, b, (short)0, cacc0, false, false);
            else
                cacc1 = __builtin_amdgcn_wmma_f32_16x16x32_f16(
                            false, a, false, b, (short)0, cacc1, false, false);
        }
    }

    float* ctxb = ctx + qkvBase + (size_t)q0 * HDIM;
    #pragma unroll
    for (int r = 0; r < 8; ++r) {
        const int row = r + rowH;
        __builtin_nontemporal_store(cacc0[r] + cacc1[r],
                                    &ctxb[(size_t)row * HDIM + wid * 16 + nIdx]);
    }
}

extern "C" void kernel_launch(void* const* d_in, const int* in_sizes, int n_in,
                              void* d_out, int out_size, void* d_ws, size_t ws_size,
                              hipStream_t stream) {
    const float* Q = (const float*)d_in[0];
    const float* K = (const float*)d_in[1];
    const float* V = (const float*)d_in[2];
    const unsigned char* mask = (const unsigned char*)d_in[3];  // jnp.bool_ = 1 byte
    const float* other = (const float*)d_in[4];

    float* ctx  = (float*)d_out;
    float* attn = ctx + (size_t)BATCH * HEADS * SEQ * HDIM;     // outputs concatenated

    const size_t shmem = (size_t)16 * SROW * sizeof(float)      // score strip
                       + (size_t)16 * HDIM * sizeof(_Float16)   // Q tile
                       + (16 * 8 + 16 + 16) * sizeof(float);    // reductions

    dim3 grid(SEQ / 16, BATCH * HEADS);
    dim3 block(128);
    sdpa_wmma_kernel<<<grid, block, shmem, stream>>>(Q, K, V, mask, other, ctx, attn);
}
